// MultiheadAttention_4707284157081
// MI455X (gfx1250) — matrix operand order repaired and checked by reading, not compiled, not run on hardware
//
#include <hip/hip_runtime.h>

// ---------------------------------------------------------------------------
// MultiheadAttention (B=4, S=4096, D=1024, single head) for gfx1250 / MI455X.
//
// Pipeline (f16 storage, f32 WMMA accumulation):
//   0a. query/key/value fp32 -> f16
//   0b. Wh,Wk,Wv,Wo fp32 [K][N] -> f16 [N][K]  (WMMA B layout)
//   1. hq/hk/hv = x @ Wh + bh          2. k2 = hk@Wk+bk ; v2 = hv@Wv+bv
//   3. P = (hq @ k2^T)/1024  (f16)     4. P = softmax_rows(P) in-place
//   5. ao = P @ v2 (v2 transposed)     6. out = ao @ Wo + bo (fp32)
//
// GEMM: 128x128 block tile, BK=64, 4 waves (128 thr), wave tile 64x64
//   (4x4 fragments -> 16 WMMAs per 16 ds_load_b128, 4x frag reuse),
//   double-buffered LDS filled by GLOBAL_LOAD_ASYNC_TO_LDS_B128 (ASYNCcnt),
//   last iteration peeled for a branch-free steady state.
// All dims are multiples of tile sizes -> no bounds checks.
// ---------------------------------------------------------------------------

typedef __attribute__((ext_vector_type(16))) _Float16 v16h;
typedef __attribute__((ext_vector_type(8)))  _Float16 v8h;
typedef __attribute__((ext_vector_type(8)))  float    v8f;

union HV16 { v16h v; v8h h[2]; };

#define BM 128
#define BN 128
#define BK 64

// Async DMA: 16B global -> LDS, tracked by ASYNCcnt (no VGPR staging).
__device__ __forceinline__ void async_copy_b128(void* lds_ptr, const void* gptr)
{
    const unsigned lds_off = (unsigned)(unsigned long long)lds_ptr; // aperture low bits
    asm volatile("global_load_async_to_lds_b128 %0, %1, off"
                 :: "v"(lds_off), "v"((unsigned long long)gptr)
                 : "memory");
}
__device__ __forceinline__ void wait_async_le16()
{ asm volatile("s_wait_asynccnt 0x10" ::: "memory"); }
__device__ __forceinline__ void wait_async_le0()
{ asm volatile("s_wait_asynccnt 0x0" ::: "memory"); }

// ---------------------------------------------------------------------------
// C[m,n] = scale * sum_k A[m,k]*BT[n,k] + bias[n];  A,BT f16; C f32 or f16.
// 128 threads = 4 waves in a 2x2 grid of 64x64 wave tiles.
// ---------------------------------------------------------------------------
template <bool OUT_F32>
__global__ __launch_bounds__(128)
void gemm_wmma_f16(const _Float16* __restrict__ A,
                   const _Float16* __restrict__ BT,
                   const float* __restrict__ bias, void* __restrict__ Cout,
                   int M, int N, int K,
                   long strideA, long strideB, long strideC, float scale)
{
    __shared__ __align__(128) _Float16 As[2][BM * BK];  // 2 x 16 KB
    __shared__ __align__(128) _Float16 Bs[2][BN * BK];  // 2 x 16 KB

    const int tid   = threadIdx.x;
    const int lane  = tid & 31;
    const int wave  = tid >> 5;
    const int waveM = wave >> 1;         // 2 x 64 rows
    const int waveN = wave & 1;          // 2 x 64 cols
    const int lmod  = lane & 15;
    const int lhalf = (lane < 16) ? 0 : 1;

    const int m0 = blockIdx.y * BM;
    const int n0 = blockIdx.x * BN;
    const long b = blockIdx.z;
    A  += b * strideA;
    BT += b * strideB;

    v8f acc[4][4];
    {
        v8f z = {};
        #pragma unroll
        for (int i = 0; i < 4; ++i)
            #pragma unroll
            for (int j = 0; j < 4; ++j) acc[i][j] = z;
    }

    // DMA slice: 8 threads/row * 8 halves; 16 rows/pass, 8 passes per tile
    const int dr = tid >> 3;
    const int dc = (tid & 7) * 8;
    auto issue_tile = [&](int kt, int buf) {
        #pragma unroll
        for (int it = 0; it < 8; ++it) {
            const int row = dr + it * 16;
            async_copy_b128(&As[buf][row * BK + dc],
                            &A[(long)(m0 + row) * K + kt + dc]);
        }
        #pragma unroll
        for (int it = 0; it < 8; ++it) {
            const int row = dr + it * 16;
            async_copy_b128(&Bs[buf][row * BK + dc],
                            &BT[(long)(n0 + row) * K + kt + dc]);
        }
    };

    auto compute_tile = [&](int buf) {
        #pragma unroll
        for (int ks = 0; ks < BK; ks += 32) {
            // A 16x32 f16 frag: lanes 0-15 K={0..15}, 16-31 K={16..31}
            v16h afrag[4];
            #pragma unroll
            for (int i = 0; i < 4; ++i) {
                const int row = waveM * 64 + i * 16 + lmod;
                const int kb  = ks + lhalf * 16;
                HV16 u;
                u.h[0] = *(const v8h*)&As[buf][row * BK + kb];
                u.h[1] = *(const v8h*)&As[buf][row * BK + kb + 8];
                afrag[i] = u.v;
            }
            // B 32x16 f16 frag: lane = col (+16 => K upper half), 16 contig K
            v16h bfrag[4];
            #pragma unroll
            for (int j = 0; j < 4; ++j) {
                const int col = waveN * 64 + j * 16 + lmod;
                bfrag[j] = *(const v16h*)&Bs[buf][col * BK + ks + lhalf * 16];
            }
            #pragma unroll
            for (int i = 0; i < 4; ++i)
                #pragma unroll
                for (int j = 0; j < 4; ++j)
                    acc[i][j] = __builtin_amdgcn_wmma_f32_16x16x32_f16(
                        false, afrag[i], false, bfrag[j],
                        (short)0, acc[i][j], false, false);
        }
    };

    const int nt = K / BK;               // >= 2 for all our shapes
    issue_tile(0, 0);
    for (int t = 0; t < nt - 1; ++t) {
        const int buf = t & 1;
        issue_tile((t + 1) * BK, buf ^ 1);  // prefetch next tile
        wait_async_le16();                  // 16 in flight = next tile only
        __syncthreads();                    // all waves' DMA visible
        compute_tile(buf);
        __syncthreads();                    // buffer free for next DMA
    }
    wait_async_le0();
    __syncthreads();
    compute_tile((nt - 1) & 1);

    // epilogue: C 16x16 f32 layout (VGPR v: M=v+8*lhalf, N=lmod)
    float*    Cf = (float*)Cout + b * strideC;
    _Float16* Ch = (_Float16*)Cout + b * strideC;
    #pragma unroll
    for (int i = 0; i < 4; ++i) {
        #pragma unroll
        for (int j = 0; j < 4; ++j) {
            const int col = n0 + waveN * 64 + j * 16 + lmod;
            const float bv = bias ? bias[col] : 0.0f;
            const int rbase = m0 + waveM * 64 + i * 16 + lhalf * 8;
            #pragma unroll
            for (int v = 0; v < 8; ++v) {
                const float r = acc[i][j][v] * scale + bv;
                const long idx = (long)(rbase + v) * N + col;
                if constexpr (OUT_F32) Cf[idx] = r;
                else                   Ch[idx] = (_Float16)r;
            }
        }
    }
}

// ---------------------------------------------------------------------------
__global__ __launch_bounds__(256)
void convert_f32_to_f16(const float* __restrict__ in,
                        _Float16* __restrict__ out, long n4)
{
    const long i = ((long)blockIdx.x * 256 + threadIdx.x) * 4;
    if (i < n4 * 4) {
        const float4 f = *(const float4*)&in[i];
        union { _Float16 h[4]; uint2 u; } pk;
        pk.h[0] = (_Float16)f.x; pk.h[1] = (_Float16)f.y;
        pk.h[2] = (_Float16)f.z; pk.h[3] = (_Float16)f.w;
        *(uint2*)&out[i] = pk.u;
    }
}

__global__ __launch_bounds__(256)
void transpose_f32_to_f16(const float* __restrict__ in,
                          _Float16* __restrict__ out, int rows, int cols)
{
    __shared__ float tile[32][33];
    const int c  = blockIdx.x * 32 + threadIdx.x;
    const int r0 = blockIdx.y * 32;
    for (int i = threadIdx.y; i < 32; i += 8)
        tile[i][threadIdx.x] = in[(long)(r0 + i) * cols + c];
    __syncthreads();
    const int c0 = blockIdx.x * 32;
    for (int i = threadIdx.y; i < 32; i += 8)
        out[(long)(c0 + i) * rows + r0 + threadIdx.x] =
            (_Float16)tile[threadIdx.x][i];
}

__global__ __launch_bounds__(256)
void transpose_f16(const _Float16* __restrict__ in,
                   _Float16* __restrict__ out, int rows, int cols)
{
    __shared__ _Float16 tile[32][33];
    const long bo = (long)blockIdx.z * rows * cols;
    const int c  = blockIdx.x * 32 + threadIdx.x;
    const int r0 = blockIdx.y * 32;
    for (int i = threadIdx.y; i < 32; i += 8)
        tile[i][threadIdx.x] = in[bo + (long)(r0 + i) * cols + c];
    __syncthreads();
    const int c0 = blockIdx.x * 32;
    for (int i = threadIdx.y; i < 32; i += 8)
        out[bo + (long)(c0 + i) * rows + r0 + threadIdx.x] =
            tile[threadIdx.x][i];
}

// In-place row softmax on f16 data (stats in fp32). 1 block per row.
__global__ __launch_bounds__(256)
void softmax_rows_f16(_Float16* __restrict__ P, int ncols)
{
    __shared__ float red[256];
    const long row = blockIdx.x;
    _Float16* p = P + row * (long)ncols;
    const int tid = threadIdx.x;

    float m = -3.402823466e38f;
    for (int c = tid; c < ncols; c += 256) m = fmaxf(m, (float)p[c]);
    red[tid] = m; __syncthreads();
    for (int s = 128; s > 0; s >>= 1) {
        if (tid < s) red[tid] = fmaxf(red[tid], red[tid + s]);
        __syncthreads();
    }
    m = red[0]; __syncthreads();

    float sum = 0.0f;
    for (int c = tid; c < ncols; c += 256) sum += __expf((float)p[c] - m);
    red[tid] = sum; __syncthreads();
    for (int s = 128; s > 0; s >>= 1) {
        if (tid < s) red[tid] += red[tid + s];
        __syncthreads();
    }
    const float inv = 1.0f / red[0];
    for (int c = tid; c < ncols; c += 256)
        p[c] = (_Float16)(__expf((float)p[c] - m) * inv);
}

// ---------------------------------------------------------------------------
extern "C" void kernel_launch(void* const* d_in, const int* in_sizes, int n_in,
                              void* d_out, int out_size, void* d_ws,
                              size_t ws_size, hipStream_t stream)
{
    (void)in_sizes; (void)n_in; (void)out_size; (void)ws_size;

    constexpr int Bb = 4, S = 4096, D = 1024;
    constexpr int M = Bb * S;

    const float* query = (const float*)d_in[0];
    const float* key   = (const float*)d_in[1];
    const float* value = (const float*)d_in[2];
    const float* Wh = (const float*)d_in[3];  const float* bh = (const float*)d_in[4];
    const float* Wk = (const float*)d_in[5];  const float* bk = (const float*)d_in[6];
    const float* Wv = (const float*)d_in[7];  const float* bv = (const float*)d_in[8];
    const float* Wo = (const float*)d_in[9];  const float* bo = (const float*)d_in[10];

    char* ws = (char*)d_ws;
    size_t off = 0;
    auto carve = [&](size_t bytes) -> void* {
        void* p = ws + off;
        off += (bytes + 255) & ~(size_t)255;
        return p;
    };
    const size_t wBytes = (size_t)D * D * sizeof(_Float16);   // 2 MB
    const size_t actB   = (size_t)M * D * sizeof(_Float16);   // 32 MB
    _Float16* WhT = (_Float16*)carve(wBytes);
    _Float16* WkT = (_Float16*)carve(wBytes);
    _Float16* WvT = (_Float16*)carve(wBytes);
    _Float16* WoT = (_Float16*)carve(wBytes);
    _Float16* qh16 = (_Float16*)carve(actB);
    _Float16* kh16 = (_Float16*)carve(actB);
    _Float16* vh16 = (_Float16*)carve(actB);
    _Float16* hq  = (_Float16*)carve(actB);
    _Float16* hk  = (_Float16*)carve(actB);
    _Float16* hv  = (_Float16*)carve(actB);
    _Float16* k2  = (_Float16*)carve(actB);
    _Float16* v2  = (_Float16*)carve(actB);
    _Float16* v2T = (_Float16*)carve(actB);
    _Float16* ao  = (_Float16*)carve(actB);
    _Float16* P   = (_Float16*)carve((size_t)Bb * S * S * sizeof(_Float16)); // 128 MB

    const dim3 blkG(128);   // GEMM blocks: 4 waves
    const dim3 blk(256);
    const dim3 tb(32, 8);
    const long nElem = (long)M * D;

    // 0a) q/k/v fp32 -> f16
    {
        const dim3 g((unsigned)(nElem / 4 / 256));
        convert_f32_to_f16<<<g, blk, 0, stream>>>(query, qh16, nElem / 4);
        convert_f32_to_f16<<<g, blk, 0, stream>>>(key,   kh16, nElem / 4);
        convert_f32_to_f16<<<g, blk, 0, stream>>>(value, vh16, nElem / 4);
    }
    // 0b) weights -> f16, transposed [N][K]
    {
        const dim3 g(D / 32, D / 32, 1);
        transpose_f32_to_f16<<<g, tb, 0, stream>>>(Wh, WhT, D, D);
        transpose_f32_to_f16<<<g, tb, 0, stream>>>(Wk, WkT, D, D);
        transpose_f32_to_f16<<<g, tb, 0, stream>>>(Wv, WvT, D, D);
        transpose_f32_to_f16<<<g, tb, 0, stream>>>(Wo, WoT, D, D);
    }
    // 1) head projection on q,k,v
    {
        const dim3 g(D / BN, M / BM, 1);
        gemm_wmma_f16<false><<<g, blkG, 0, stream>>>(qh16, WhT, bh, hq, M, D, D, 0, 0, 0, 1.0f);
        gemm_wmma_f16<false><<<g, blkG, 0, stream>>>(kh16, WhT, bh, hk, M, D, D, 0, 0, 0, 1.0f);
        gemm_wmma_f16<false><<<g, blkG, 0, stream>>>(vh16, WhT, bh, hv, M, D, D, 0, 0, 0, 1.0f);
    }
    // 2) k / v projections
    {
        const dim3 g(D / BN, M / BM, 1);
        gemm_wmma_f16<false><<<g, blkG, 0, stream>>>(hk, WkT, bk, k2, M, D, D, 0, 0, 0, 1.0f);
        gemm_wmma_f16<false><<<g, blkG, 0, stream>>>(hv, WvT, bv, v2, M, D, D, 0, 0, 0, 1.0f);
    }
    // 3) P = hq @ k2^T / 1024  (k2 row-major IS B^T layout [j][d]; f16 out)
    {
        const dim3 g(S / BN, S / BM, Bb);
        gemm_wmma_f16<false><<<g, blkG, 0, stream>>>(
            hq, k2, nullptr, P, S, S, D,
            (long)S * D, (long)S * D, (long)S * S, 1.0f / (float)D);
    }
    // 4) softmax rows, in place on P
    softmax_rows_f16<<<dim3(M), blk, 0, stream>>>(P, S);
    // 5) ao = P @ v2  (stage v2^T as B)
    {
        const dim3 gt(D / 32, S / 32, Bb);
        transpose_f16<<<gt, tb, 0, stream>>>(v2, v2T, S, D);
        const dim3 g(D / BN, S / BM, Bb);
        gemm_wmma_f16<false><<<g, blkG, 0, stream>>>(
            P, v2T, nullptr, ao, S, D, S,
            (long)S * S, (long)D * S, (long)S * D, 1.0f);
    }
    // 6) out = ao @ Wo + bo
    {
        const dim3 g(D / BN, M / BM, 1);
        gemm_wmma_f16<true><<<g, blkG, 0, stream>>>(
            ao, WoT, bo, (float*)d_out, M, D, D, 0, 0, 0, 1.0f);
    }
}
